// LinearAttention1d_25305947308725
// MI455X (gfx1250) — matrix operand order repaired and checked by reading, not compiled, not run on hardware
//
#include <hip/hip_runtime.h>
#include <hip/hip_bf16.h>
#include <hip/hip_fp16.h>

// ---------------------------------------------------------------------------
// LinearAttention1d for MI455X (gfx1250, wave32, WMMA 16x16x32 f16)
//
// Reference collapses to:
//   scale[b,c]  = (sum_t softmax_d(k)[c,t]) * (sum_t v[c,t]) / (sum_t exp(q[c,t]))
//   out[b,o,t]  = sum_c (w_out[o,c]*scale[b,c]) * exp(q[b,c,t]) + b_out[o]
// and  sum_t v[c,t] = Wv @ (sum_t x[:,t])   (linearity -> no v GEMM needed)
// ---------------------------------------------------------------------------

#define BATCH   8
#define DIM     64
#define TLEN    16384
#define INNER   128          // heads*dim_head = 4*32
#define QKVROWS 384          // 3*INNER
#define TT      256          // time tile per block
#define NCHUNK  (TLEN / TT)  // 64
#define SA      72           // LDS row stride (halves), K=64 mats: 144B, 16B aligned, bank-staggered
#define SE      136          // LDS row stride (halves), K=128 mats: 272B, 16B aligned

typedef __attribute__((ext_vector_type(2)))  _Float16 v2h;
typedef __attribute__((ext_vector_type(8)))  _Float16 v8h;
typedef __attribute__((ext_vector_type(16))) _Float16 v16h;
typedef __attribute__((ext_vector_type(8)))  float    v8f;

// ---- WMMA fragment helpers (CDNA5 VGPR layouts, CDNA5 ISA) -----
__device__ __forceinline__ v16h load_frag_a(const _Float16* __restrict__ S,
                                            int stride, int mbase, int kb, int lane) {
  const int hi  = (lane >> 4) & 1;
  const int row = mbase + (lane & 15);
  const _Float16* p = S + row * stride + kb + hi * 8;
  v8h lo = *(const v8h*)(p);
  v8h hv = *(const v8h*)(p + 16);
  return __builtin_shufflevector(lo, hv, 0,1,2,3,4,5,6,7,8,9,10,11,12,13,14,15);
}

__device__ __forceinline__ v16h load_frag_b(const _Float16* __restrict__ S,
                                            int stride, int nbase, int kb, int lane) {
  const int hi  = (lane >> 4) & 1;
  const int col = nbase + (lane & 15);
  const _Float16* p = S + col * stride + kb + hi * 8;
  v8h lo = *(const v8h*)(p);
  v8h hv = *(const v8h*)(p + 16);
  return __builtin_shufflevector(lo, hv, 0,1,2,3,4,5,6,7,8,9,10,11,12,13,14,15);
}

__device__ __forceinline__ v8f wmma16(v16h a, v16h b, v8f c) {
  return __builtin_amdgcn_wmma_f32_16x16x32_f16(false, a, false, b, (short)0, c, false, false);
}

// ---------------------------------------------------------------------------
// Kernel 1: per-chunk stats.
//   stats[b][chunk][0:128]   = sum_t exp(q)
//   stats[b][chunk][128:256] = sum_t softmax_d(k)
//   stats[b][chunk][256:384] = sum_t v   (= Wv @ chunk-xsum)
// Row sums use the C-layout fact that 16 lanes of a half-group hold 16 t's of
// one row: a single unconditional LDS atomic add per D register, no shuffles.
// ---------------------------------------------------------------------------
__global__ __launch_bounds__(256) void
la1d_stats(const float* __restrict__ x, const float* __restrict__ wqkv,
           float* __restrict__ stats) {
  extern __shared__ char smem[];
  _Float16* Ash  = (_Float16*)smem;                // [QKVROWS][SA]   55296 B
  _Float16* Bts  = Ash + QKVROWS * SA;             // [TT][SA]        36864 B
  float*    accw = (float*)(Bts + TT * SA);        // [8 waves][256]   8192 B (q+k rows)
  float*    xpart = accw + 8 * 256;                // [4][64]          1024 B

  const int tid  = threadIdx.x;
  const int lane = tid & 31;
  const int wave = tid >> 5;
  const int hi   = (lane >> 4) & 1;
  const int chunk = blockIdx.x;
  const int bb    = blockIdx.y;
  const int t0    = chunk * TT;

  for (int i = tid; i < QKVROWS * DIM; i += 256) {
    const int r = i >> 6, c = i & 63;
    Ash[r * SA + c] = (_Float16)wqkv[i];
  }
  const float* xb = x + (size_t)bb * DIM * TLEN + t0;
  for (int i = tid; i < DIM * TT; i += 256) {
    const int c = i >> 8, tt = i & (TT - 1);
    Bts[tt * SA + c] = (_Float16)xb[(size_t)c * TLEN + tt];
  }
  for (int i = tid; i < 8 * 256; i += 256) accw[i] = 0.f;
  __syncthreads();

  // ---- chunk x column-sum partials (for the v path), deterministic slots --
  {
    const int c = tid & 63, seg = tid >> 6;     // 4 segments of 64 t's
    float s = 0.f;
    for (int tt = seg * 64; tt < seg * 64 + 64; ++tt) s += (float)Bts[tt * SA + c];
    xpart[seg * 64 + c] = s;
  }

  float* aw = accw + wave * 256;

  for (int nt = wave * 2; nt < wave * 2 + 2; ++nt) {
    // ---- Q rows (0..127): sum_t exp(q) ------------------------------------
    for (int mt = 0; mt < 8; ++mt) {
      v8f c = {};
      #pragma unroll
      for (int ks = 0; ks < 64; ks += 32) {
        v16h a  = load_frag_a(Ash, SA, mt * 16, ks, lane);
        v16h bm = load_frag_b(Bts, SA, nt * 16, ks, lane);
        c = wmma16(a, bm, c);
      }
      #pragma unroll
      for (int r = 0; r < 8; ++r)
        atomicAdd(&aw[mt * 16 + r + hi * 8], __expf(c[r]));   // ds_add_f32
    }
    // ---- K rows (128..255): softmax over d (32 rows / head), then sum_t ---
    for (int h = 0; h < 4; ++h) {
      const int mtA = 8 + 2 * h;
      v8f c0 = {}, c1 = {};
      #pragma unroll
      for (int ks = 0; ks < 64; ks += 32) {
        v16h bm = load_frag_b(Bts, SA, nt * 16, ks, lane);
        v16h a0 = load_frag_a(Ash, SA, mtA * 16, ks, lane);
        c0 = wmma16(a0, bm, c0);
        v16h a1 = load_frag_a(Ash, SA, (mtA + 1) * 16, ks, lane);
        c1 = wmma16(a1, bm, c1);
      }
      float e0[8], e1[8], den = 0.f;
      #pragma unroll
      for (int r = 0; r < 8; ++r) {
        e0[r] = __expf(c0[r]); e1[r] = __expf(c1[r]);
        den += e0[r] + e1[r];
      }
      den += __shfl_xor(den, 16, 32);   // lane^16 holds the head's other 16 rows
      const float inv = 1.0f / den;
      #pragma unroll
      for (int r = 0; r < 8; ++r) {
        atomicAdd(&aw[128 + (2 * h) * 16 + r + hi * 8],     e0[r] * inv);
        atomicAdd(&aw[128 + (2 * h + 1) * 16 + r + hi * 8], e1[r] * inv);
      }
    }
  }
  __syncthreads();

  float* outp = stats + ((size_t)bb * NCHUNK + chunk) * QKVROWS;
  // q+k rows: reduce the 8 per-wave slices
  {
    float s = 0.f;
    #pragma unroll
    for (int w = 0; w < 8; ++w) s += accw[w * 256 + tid];
    outp[tid & 255] = s;   // tid in [0,256)
  }
  if (tid < 64) xpart[tid] += xpart[64 + tid] + xpart[128 + tid] + xpart[192 + tid];
  __syncthreads();
  // v rows: vsum_chunk[row] = dot(Wv[row,:], xsum_chunk)
  if (tid < 128) {
    const _Float16* wv = Ash + (2 * INNER + tid) * SA;
    float v = 0.f;
    #pragma unroll
    for (int c = 0; c < DIM; ++c) v += (float)wv[c] * xpart[c];
    outp[2 * INNER + tid] = v;
  }
}

// ---------------------------------------------------------------------------
// Kernel 2: reduce chunks -> scale[b,c] = ksum*vsum/qsum
// ---------------------------------------------------------------------------
__global__ __launch_bounds__(256) void
la1d_scale(const float* __restrict__ stats, float* __restrict__ scale) {
  const int idx = blockIdx.x * 256 + threadIdx.x;
  if (idx >= BATCH * INNER) return;
  const int bb = idx >> 7, c = idx & 127;
  const float* p = stats + (size_t)bb * NCHUNK * QKVROWS;
  float qs = 0.f, ks = 0.f, vs = 0.f;
  for (int ch = 0; ch < NCHUNK; ++ch) {
    qs += p[ch * QKVROWS + c];
    ks += p[ch * QKVROWS + INNER + c];
    vs += p[ch * QKVROWS + 2 * INNER + c];
  }
  scale[idx] = ks * vs / qs;
}

// ---------------------------------------------------------------------------
// Kernel 3: out[b,o,t] = sum_c (w_out[o,c]*scale[b,c]) * exp(q[b,c,t]) + b_out[o]
// ---------------------------------------------------------------------------
__global__ __launch_bounds__(256) void
la1d_out(const float* __restrict__ x, const float* __restrict__ wqkv,
         const float* __restrict__ wout, const float* __restrict__ bout,
         const float* __restrict__ scale, float* __restrict__ out) {
  extern __shared__ char smem[];
  _Float16* Aq  = (_Float16*)smem;        // [128][SA]  18432 B (q rows of w_qkv)
  _Float16* Bts = Aq + INNER * SA;        // [TT][SA]   36864 B
  _Float16* Et  = Bts + TT * SA;          // [TT][SE]   69632 B  exp(q) as B-matrix
  _Float16* W2  = Et + TT * SE;           // [64][SE]   17408 B

  const int tid  = threadIdx.x;
  const int lane = tid & 31;
  const int wave = tid >> 5;
  const int hi   = (lane >> 4) & 1;
  const int chunk = blockIdx.x;
  const int bb    = blockIdx.y;
  const int t0    = chunk * TT;

  for (int i = tid; i < INNER * DIM; i += 256) {       // w_q rows 0..127
    const int r = i >> 6, c = i & 63;
    Aq[r * SA + c] = (_Float16)wqkv[i];
  }
  const float* xb = x + (size_t)bb * DIM * TLEN + t0;
  for (int i = tid; i < DIM * TT; i += 256) {
    const int c = i >> 8, tt = i & (TT - 1);
    Bts[tt * SA + c] = (_Float16)xb[(size_t)c * TLEN + tt];
  }
  const float* sc = scale + bb * INNER;
  for (int i = tid; i < DIM * INNER; i += 256) {       // W2 = w_out * scale
    const int o = i >> 7, c = i & 127;
    W2[o * SE + c] = (_Float16)(wout[i] * sc[c]);
  }
  __syncthreads();

  // GEMM1: q = Wq @ x, then exp -> Et[t][c] (paired-half b32 stores)
  for (int nt = wave * 2; nt < wave * 2 + 2; ++nt) {
    for (int mt = 0; mt < 8; ++mt) {
      v8f c = {};
      #pragma unroll
      for (int ks = 0; ks < 64; ks += 32) {
        v16h a  = load_frag_a(Aq, SA, mt * 16, ks, lane);
        v16h bm = load_frag_b(Bts, SA, nt * 16, ks, lane);
        c = wmma16(a, bm, c);
      }
      _Float16* ep = Et + (nt * 16 + (lane & 15)) * SE + mt * 16 + hi * 8;
      #pragma unroll
      for (int r = 0; r < 8; r += 2) {
        v2h pk = { (_Float16)__expf(c[r]), (_Float16)__expf(c[r + 1]) };
        *(v2h*)(ep + r) = pk;                          // 4B-aligned ds_store_b32
      }
    }
  }
  __syncthreads();

  // GEMM2: out = W2 @ exp(q) + b_out   (M=64, K=128)
  float* ob = out + (size_t)bb * DIM * TLEN + t0;
  for (int nt = wave * 2; nt < wave * 2 + 2; ++nt) {
    for (int mt = 0; mt < 4; ++mt) {
      v8f c = {};
      #pragma unroll
      for (int ks = 0; ks < 128; ks += 32) {
        v16h a  = load_frag_a(W2, SE, mt * 16, ks, lane);
        v16h bm = load_frag_b(Et, SE, nt * 16, ks, lane);
        c = wmma16(a, bm, c);
      }
      #pragma unroll
      for (int r = 0; r < 8; ++r) {
        const int o = mt * 16 + r + hi * 8;
        ob[(size_t)o * TLEN + nt * 16 + (lane & 15)] = c[r] + bout[o];
      }
    }
  }
}

// ---------------------------------------------------------------------------
extern "C" void kernel_launch(void* const* d_in, const int* in_sizes, int n_in,
                              void* d_out, int out_size, void* d_ws, size_t ws_size,
                              hipStream_t stream) {
  const float* x    = (const float*)d_in[0];
  const float* wqkv = (const float*)d_in[1];
  const float* wout = (const float*)d_in[2];
  const float* bout = (const float*)d_in[3];
  float* out   = (float*)d_out;
  float* stats = (float*)d_ws;                                // [8][64][384]
  float* scale = stats + (size_t)BATCH * NCHUNK * QKVROWS;    // [8][128]

  const size_t lds1 = (size_t)(QKVROWS * SA + TT * SA) * sizeof(_Float16)
                    + (8 * 256 + 4 * 64) * sizeof(float);     // ~100 KB
  const size_t lds3 = (size_t)(INNER * SA + TT * SA + TT * SE + DIM * SE)
                    * sizeof(_Float16);                       // ~139 KB

  la1d_stats<<<dim3(NCHUNK, BATCH), 256, lds1, stream>>>(x, wqkv, stats);
  la1d_scale<<<dim3(4), 256, 0, stream>>>(stats, scale);
  la1d_out<<<dim3(NCHUNK, BATCH), 256, lds3, stream>>>(x, wqkv, wout, bout, scale, out);
}
